// ModulatedConv2d_1726576856945
// MI455X (gfx1250) — matrix operand order repaired and checked by reading, not compiled, not run on hardware
//
#include <hip/hip_runtime.h>
#include <hip/hip_bf16.h>

typedef __attribute__((ext_vector_type(16))) __bf16 v16bf;
typedef __attribute__((ext_vector_type(8)))  __bf16 v8bf;
typedef __attribute__((ext_vector_type(8)))  float  v8f;
typedef __attribute__((ext_vector_type(4)))  int    v4i;
typedef v4i __attribute__((address_space(1)))* gv4i_p;   // global int4*
typedef v4i __attribute__((address_space(3)))* lv4i_p;   // LDS int4*

#define NB   16
#define CIN  512
#define COUT 512
#define HW   1024
#define PH   34          // padded plane height/width
#define PP   1156        // 34*34
#define PITCH 40         // LDS row pitch in bf16: 16B-aligned, conflict-free

// CDNA5 async global->LDS path (ASYNCcnt-tracked), guarded so we always compile.
#if defined(__has_builtin)
#if __has_builtin(__builtin_amdgcn_global_load_async_to_lds_b128) && \
    __has_builtin(__builtin_amdgcn_s_wait_asynccnt)
#define USE_ASYNC_LDS 1
#endif
#endif
#ifndef USE_ASYNC_LDS
#define USE_ASYNC_LDS 0
#endif

// ---------------- Kernel A: s = leaky_relu(style @ affine_w^T + affine_b, 0.2)
__global__ __launch_bounds__(256) void affine_kernel(
    const float* __restrict__ style, const float* __restrict__ aw,
    const float* __restrict__ ab, float* __restrict__ s) {
  int idx = blockIdx.x * 256 + threadIdx.x;
  if (idx >= NB * CIN) return;
  int b = idx >> 9, ci = idx & 511;
  const float* st = style + b * 512;
  const float* wr = aw + (size_t)ci * 512;
  float acc = ab[ci];
  for (int j = 0; j < 512; ++j) acc = fmaf(st[j], wr[j], acc);
  s[idx] = acc > 0.f ? acc : 0.2f * acc;
}

// ---------------- Kernel B1: wsq[co,ci] = sum_k weight[co,ci,k]^2
__global__ __launch_bounds__(256) void wsq_kernel(
    const float* __restrict__ w, float* __restrict__ wsq) {
  int idx = blockIdx.x * 256 + threadIdx.x;
  if (idx >= COUT * CIN) return;
  const float* p = w + (size_t)idx * 9;
  float a = 0.f;
#pragma unroll
  for (int k = 0; k < 9; ++k) { float v = p[k]; a = fmaf(v, v, a); }
  wsq[idx] = a;
}

// ---------------- Kernel B2: d[b,co] = rsqrt(sum_ci wsq[co,ci]*s[b,ci]^2 + eps)
__global__ __launch_bounds__(256) void demod_kernel(
    const float* __restrict__ wsq, const float* __restrict__ s,
    float* __restrict__ dmod) {
  int idx = blockIdx.x * 256 + threadIdx.x;
  if (idx >= NB * COUT) return;
  int b = idx >> 9, co = idx & 511;
  const float* wr = wsq + (size_t)co * 512;
  const float* sr = s + b * 512;
  float a = 0.f;
  for (int ci = 0; ci < 512; ++ci) { float sv = sr[ci]; a = fmaf(wr[ci] * sv, sv, a); }
  dmod[idx] = rsqrtf(a + 1e-8f);
}

// ---------------- Kernel C: xpad[b,ci,pr,pc] = bf16(x*s) with 1-pixel zero border
__global__ __launch_bounds__(256) void xpad_kernel(
    const float* __restrict__ x, const float* __restrict__ s,
    __bf16* __restrict__ xp) {
  size_t i = (size_t)blockIdx.x * 256 + threadIdx.x;   // 16*512*1156
  if (i >= (size_t)NB * CIN * PP) return;
  int sp = (int)(i % PP);
  size_t bci = i / PP;
  int pr = sp / PH, pc = sp % PH;
  float v = 0.f;
  if (pr >= 1 && pr <= 32 && pc >= 1 && pc <= 32)
    v = x[bci * 1024 + (size_t)(pr - 1) * 32 + (pc - 1)] * s[bci];
  xp[i] = (__bf16)v;
}

// ---------------- Kernel D: wtb[k9][co][ci] = bf16(weight[co,ci,k9])
__global__ __launch_bounds__(256) void wtb_kernel(
    const float* __restrict__ w, __bf16* __restrict__ wtb) {
  int i = blockIdx.x * 256 + threadIdx.x;
  if (i >= 9 * COUT * CIN) return;
  int ci = i & 511;
  int co = (i >> 9) & 511;
  int k9 = i >> 18;
  wtb[i] = (__bf16)(w[((size_t)(co * 512 + ci)) * 9 + k9]);
}

// ---------------- Kernel E: implicit-GEMM conv, 36 WMMAs per wave per ci-chunk
// Block tile 64 (Cout) x 128 (HW = 4 output rows). 8 waves: 2(M) x 4(N),
// wave tile 32x32 = four 16x16 f32 accumulators.
__global__ __launch_bounds__(256) void conv_wmma_kernel(
    const __bf16* __restrict__ xpad,   // [B][Cin][34][34]
    const __bf16* __restrict__ wtb,    // [9][Cout][Cin]
    const float*  __restrict__ dmod,   // [B][Cout]
    float*        __restrict__ out) {  // [B][Cout][HW]
  __shared__ __bf16 As[9][64][PITCH];   // weights, all 9 taps for this ci-chunk
  __shared__ __bf16 Bsp[204][PITCH];    // [spatial 6x34][ci] transposed stage
  __shared__ float  dsh[64];

  const int b   = blockIdx.z;
  const int m0  = blockIdx.y * 64;      // Cout base
  const int n0  = blockIdx.x * 128;     // HW base (4 output rows)
  const int oh0 = n0 >> 5;              // first staged padded row
  const int t    = threadIdx.x;
  const int lane = t & 31;
  const int wv   = t >> 5;
  const int wm   = wv & 1;              // M sub (x32)
  const int wn   = wv >> 1;             // N sub (x32)
  const int lhalf = lane >> 4;
  const int l16   = lane & 15;

  if (t < 64) dsh[t] = dmod[b * COUT + m0 + t];

  v8f acc[2][2] = {};

  const int arow = t >> 2;              // 0..63
  const int acol = (t & 3) << 3;        // 0,8,16,24
  const int bci_l = t >> 3;             // 0..31
  const int bpart = t & 7;              // 8 threads per ci

  for (int cic = 0; cic < CIN / 32; ++cic) {
    __syncthreads();   // protect LDS reuse before new stores land
    // ---- stage A: wtb[k9][m0+row][cic*32+col] for all 9 taps
    {
      const __bf16* src = wtb + ((size_t)(m0 + arow)) * CIN + cic * 32 + acol;
#if USE_ASYNC_LDS
#pragma unroll
      for (int k9 = 0; k9 < 9; ++k9) {
        __builtin_amdgcn_global_load_async_to_lds_b128(
            (gv4i_p)(src + (size_t)k9 * COUT * CIN),
            (lv4i_p)(&As[k9][arow][acol]),
            0, 0);
      }
#else
#pragma unroll
      for (int k9 = 0; k9 < 9; ++k9)
        *(v8bf*)(&As[k9][arow][acol]) =
            *(const v8bf*)(src + (size_t)k9 * COUT * CIN);
#endif
      __builtin_prefetch(src + 32, 0, 0);  // next ci-chunk -> global_prefetch
    }
    // ---- stage B: 6 padded rows x 34 cols x 32 ci, transposed, no bounds checks
    // (overlaps the in-flight async A transfers)
    {
      const __bf16* src =
          xpad + ((size_t)(b * CIN + cic * 32 + bci_l)) * PP + oh0 * PH;
#pragma unroll
      for (int e = 0; e < 26; ++e) {
        int sp = bpart * 26 + e;
        if (sp < 204) Bsp[sp][bci_l] = src[sp];
      }
    }
#if USE_ASYNC_LDS
    __builtin_amdgcn_s_wait_asynccnt(0);   // A tile landed in LDS
#endif
    __syncthreads();

#pragma unroll
    for (int k9 = 0; k9 < 9; ++k9) {
      const int dh = k9 / 3, dw = k9 % 3;   // 0..2 (halo folded into padding)
      v16bf a0, a1, bf0, bf1;
      // A fragments (16x32): lanes 0-15 K={0..7,16..23}; lanes 16-31 +8
      {
        const int kb = lhalf << 3;
        const int r0 = wm * 32 + l16;
        ((v8bf*)&a0)[0] = *(const v8bf*)(&As[k9][r0][kb]);
        ((v8bf*)&a0)[1] = *(const v8bf*)(&As[k9][r0][kb + 16]);
        ((v8bf*)&a1)[0] = *(const v8bf*)(&As[k9][r0 + 16][kb]);
        ((v8bf*)&a1)[1] = *(const v8bf*)(&As[k9][r0 + 16][kb + 16]);
      }
      // B fragments (32x16): lanes 0-15 K={0..7,16..23}; lanes 16-31 +8
      {
        const int kb = lhalf << 3;
        const int np0 = wn * 32 + l16;
        const int np1 = np0 + 16;
        const int sp0 = ((np0 >> 5) + dh) * PH + (np0 & 31) + dw;
        const int sp1 = ((np1 >> 5) + dh) * PH + (np1 & 31) + dw;
        ((v8bf*)&bf0)[0] = *(const v8bf*)(&Bsp[sp0][kb]);
        ((v8bf*)&bf0)[1] = *(const v8bf*)(&Bsp[sp0][kb + 16]);
        ((v8bf*)&bf1)[0] = *(const v8bf*)(&Bsp[sp1][kb]);
        ((v8bf*)&bf1)[1] = *(const v8bf*)(&Bsp[sp1][kb + 16]);
      }
      acc[0][0] = __builtin_amdgcn_wmma_f32_16x16x32_bf16(
          false, a0, false, bf0, (short)0, acc[0][0], false, false);
      acc[0][1] = __builtin_amdgcn_wmma_f32_16x16x32_bf16(
          false, a0, false, bf1, (short)0, acc[0][1], false, false);
      acc[1][0] = __builtin_amdgcn_wmma_f32_16x16x32_bf16(
          false, a1, false, bf0, (short)0, acc[1][0], false, false);
      acc[1][1] = __builtin_amdgcn_wmma_f32_16x16x32_bf16(
          false, a1, false, bf1, (short)0, acc[1][1], false, false);
    }
  }

  __syncthreads();
  // ---- epilogue: C layout VGPR r: lanes0-15 M=r,N=lane; lanes16-31 M=r+8
  const int hwbase = n0 + wn * 32 + l16;
#pragma unroll
  for (int mi = 0; mi < 2; ++mi) {
    const int co_base = m0 + wm * 32 + mi * 16 + lhalf * 8;
#pragma unroll
    for (int r = 0; r < 8; ++r) {
      int co = co_base + r;
      float dm = dsh[co - m0];
      float* orow = out + ((size_t)(b * COUT + co)) * HW;
      orow[hwbase]      = acc[mi][0][r] * dm;
      orow[hwbase + 16] = acc[mi][1][r] * dm;
    }
  }
}

extern "C" void kernel_launch(void* const* d_in, const int* in_sizes, int n_in,
                              void* d_out, int out_size, void* d_ws, size_t ws_size,
                              hipStream_t stream) {
  const float* input  = (const float*)d_in[0];   // [16,512,32,32]
  const float* style  = (const float*)d_in[1];   // [16,512]
  const float* weight = (const float*)d_in[2];   // [512,512,3,3]
  const float* aff_w  = (const float*)d_in[3];   // [512,512]
  const float* aff_b  = (const float*)d_in[4];   // [512]
  float* out = (float*)d_out;                    // [16,512,32,32]

  char* ws = (char*)d_ws;
  float*  s_buf = (float*)(ws);                                  // 32 KB
  float*  d_buf = (float*)(ws + 32768);                          // 32 KB
  float*  wsq   = (float*)(ws + 65536);                          // 1 MB
  __bf16* wtb   = (__bf16*)(ws + 65536 + 1048576);               // 4.5 MB
  __bf16* xpad  = (__bf16*)(ws + 65536 + 1048576 + 4718592);     // 18.9 MB

  affine_kernel<<<(NB * CIN + 255) / 256, 256, 0, stream>>>(style, aff_w, aff_b, s_buf);
  wsq_kernel<<<(COUT * CIN + 255) / 256, 256, 0, stream>>>(weight, wsq);
  demod_kernel<<<(NB * COUT + 255) / 256, 256, 0, stream>>>(wsq, s_buf, d_buf);
  xpad_kernel<<<(int)(((size_t)NB * CIN * PP + 255) / 256), 256, 0, stream>>>(input, s_buf, xpad);
  wtb_kernel<<<(9 * COUT * CIN + 255) / 256, 256, 0, stream>>>(weight, wtb);

  dim3 grid(HW / 128, COUT / 64, NB);   // 8 x 8 x 16
  conv_wmma_kernel<<<grid, 256, 0, stream>>>(xpad, wtb, d_buf, out);
}
